// NeuralSpline_44985487458630
// MI455X (gfx1250) — matrix operand order repaired and checked by reading, not compiled, not run on hardware
//
#include <hip/hip_runtime.h>
#include <hip/hip_bf16.h>

typedef __attribute__((ext_vector_type(16))) __bf16 v16bf;
typedef __attribute__((ext_vector_type(8)))  float  v8f;

#define SPL_STEP (1.0f/9.0f)
#define BATCH 64
#define PIX   (256*256)
#define OUT_MAIN ((size_t)2*BATCH*3*PIX)   // 25,165,824 floats, then splines

// ---------------------------------------------------------------------------
// Implicit-GEMM conv 3x3 stride-2 VALID, fused bias + ReLU + (optional) BN.
// One wave32 computes a 16(Cout) x 16(spatial) tile, K-chunks of 32 (bf16 WMMA).
// im2col address decode is precomputed into an LDS offset table per block.
// ---------------------------------------------------------------------------
__global__ __launch_bounds__(256) void conv_wmma_kernel(
    const float* __restrict__ in, const float* __restrict__ w,
    const float* __restrict__ bias,
    const float* __restrict__ bng, const float* __restrict__ bnb,
    const float* __restrict__ bnm, const float* __restrict__ bnv,
    float* __restrict__ out,
    int Bn, int Cin, int Hin, int Win, int Cout, int Hout, int Wout,
    int has_bn)
{
    __shared__ int off_tab[576];                  // KK <= 64*9

    const int S  = Hout * Wout;
    const int KK = Cin * 9;

    // Cooperative fill of im2col offsets: off[k] = ci*Hin*Win + kh*Win + kw
    for (int k = threadIdx.x; k < KK; k += blockDim.x) {
        int ci = k / 9, r = k - ci * 9;
        int kh = r / 3,  kw = r - kh * 3;
        off_tab[k] = (ci * Hin + kh) * Win + kw;
    }
    __syncthreads();

    const int tiles_s  = (S + 15) >> 4;
    const int tiles_oc = (Cout + 15) >> 4;
    const int total    = Bn * tiles_s * tiles_oc;

    const int wave = blockIdx.x * (blockDim.x >> 5) + (threadIdx.x >> 5);
    if (wave >= total) return;                    // wave-uniform: EXEC stays full
    const int lane = threadIdx.x & 31;

    int b   = wave / (tiles_s * tiles_oc);
    int rem = wave - b * (tiles_s * tiles_oc);
    int st  = rem / tiles_oc;                     // spatial tile
    int ot  = rem - st * tiles_oc;                // out-channel tile

    // A-fragment role (weights): row M, contiguous K-halves (k = 16*half + j), same order as B
    const int  am   = lane & 15;
    const bool ahi  = lane >= 16;
    const int  aoc  = ot * 16 + am;
    const bool aval = aoc < Cout;
    const float* wrow = w + (size_t)aoc * KK;

    // B-fragment role (im2col patch): column N = spatial position
    const int  bn   = lane & 15;
    const bool bhi  = lane >= 16;
    const int  bp   = st * 16 + bn;
    const bool bval = bp < S;
    const int  boh  = bval ? (bp / Wout) : 0;
    const int  bow  = bval ? (bp - boh * Wout) : 0;
    // per-lane patch base: input pixel (2*oh, 2*ow) of this batch image
    const float* pbase = in + (size_t)b * Cin * Hin * Win
                            + (size_t)(boh * 2) * Win + (bow * 2);

    v8f acc = {};
    for (int k0 = 0; k0 < KK; k0 += 32) {
        v16bf af, bf;
        #pragma unroll
        for (int j = 0; j < 16; ++j) {
            // A element j -> K offset = j (+16 for upper lanes), matching B
            int ko = j + (ahi ? 16 : 0);
            int k  = k0 + ko;
            float v = (aval && k < KK) ? wrow[k] : 0.0f;
            af[j] = (__bf16)v;
        }
        #pragma unroll
        for (int j = 0; j < 16; ++j) {
            // B element j -> K row = k0 + j (+16 for upper lanes)
            int k = k0 + j + (bhi ? 16 : 0);
            float v = 0.0f;
            if (bval && k < KK) v = pbase[off_tab[k]];
            bf[j] = (__bf16)v;
        }
        acc = __builtin_amdgcn_wmma_f32_16x16x32_bf16(
                  false, af, false, bf, (short)0, acc, false, false);
    }

    // C/D layout: VGPR r, lane<16 -> (M=r, N=lane); lane>=16 -> (M=8+r, N=lane-16)
    const int  cn  = lane & 15;
    const bool chi = lane >= 16;
    const int  cp  = st * 16 + cn;
    #pragma unroll
    for (int r = 0; r < 8; ++r) {
        int oc = ot * 16 + r + (chi ? 8 : 0);
        if (oc < Cout && cp < S) {
            float v = acc[r] + bias[oc];
            v = fmaxf(v, 0.0f);
            if (has_bn) {
                float s = bng[oc] * rsqrtf(bnv[oc] + 1e-5f);
                v = (v - bnm[oc]) * s + bnb[oc];
            }
            out[((size_t)b * Cout + oc) * S + cp] = v;
        }
    }
}

// ---------------------------------------------------------------------------
// WMMA FC: out[M x N] = act(A[M x K] @ W[N x K]^T + bias), K multiple of 32.
// One wave per 16x16 output tile.
// ---------------------------------------------------------------------------
__global__ __launch_bounds__(256) void fc_wmma_kernel(
    const float* __restrict__ A, const float* __restrict__ W,
    const float* __restrict__ bias, float* __restrict__ out,
    int M, int N, int K, int relu)
{
    const int tiles_m = (M + 15) >> 4;
    const int tiles_n = (N + 15) >> 4;
    const int total   = tiles_m * tiles_n;

    const int wave = blockIdx.x * (blockDim.x >> 5) + (threadIdx.x >> 5);
    if (wave >= total) return;
    const int lane = threadIdx.x & 31;

    const int mt = wave / tiles_n;
    const int nt = wave - mt * tiles_n;

    const int  am   = lane & 15;
    const bool ahi  = lane >= 16;
    const int  arow = mt * 16 + am;
    const bool aval = arow < M;
    const float* arow_p = A + (size_t)arow * K;

    const int  bn   = lane & 15;
    const bool bhi  = lane >= 16;
    const int  bcol = nt * 16 + bn;
    const bool bval = bcol < N;
    const float* wcol_p = W + (size_t)bcol * K;   // B[k][n] = W[n][k]

    v8f acc = {};
    for (int k0 = 0; k0 < K; k0 += 32) {
        v16bf af, bf;
        #pragma unroll
        for (int j = 0; j < 16; ++j) {
            int k = k0 + j + (ahi ? 16 : 0);
            float v = (aval && k < K) ? arow_p[k] : 0.0f;
            af[j] = (__bf16)v;
        }
        #pragma unroll
        for (int j = 0; j < 16; ++j) {
            int k = k0 + j + (bhi ? 16 : 0);
            float v = (bval && k < K) ? wcol_p[k] : 0.0f;
            bf[j] = (__bf16)v;
        }
        acc = __builtin_amdgcn_wmma_f32_16x16x32_bf16(
                  false, af, false, bf, (short)0, acc, false, false);
    }

    const int  cn  = lane & 15;
    const bool chi = lane >= 16;
    const int  n   = nt * 16 + cn;
    #pragma unroll
    for (int r = 0; r < 8; ++r) {
        int m = mt * 16 + r + (chi ? 8 : 0);
        if (m < M && n < N) {
            float v = acc[r] + bias[n];
            if (relu) v = fmaxf(v, 0.0f);
            out[(size_t)m * N + n] = v;
        }
    }
}

// ---------------------------------------------------------------------------
// Global average pool over S spatial elements per (b, c)
// ---------------------------------------------------------------------------
__global__ void pool_kernel(const float* __restrict__ y, float* __restrict__ feat,
                            int total, int S)
{
    int t = blockIdx.x * blockDim.x + threadIdx.x;
    if (t >= total) return;
    const float* p = y + (size_t)t * S;
    float s = 0.0f;
    for (int i = 0; i < S; ++i) s += p[i];
    feat[t] = s / (float)S;
}

// ---------------------------------------------------------------------------
// Natural cubic spline coefficients via Thomas solve of tridiag(1,4,1):
// identical to MAT = (6/h^2) * vstack(0, inv(T), 0) @ D2 in the reference.
// coef layout: [(b*2+e)*3 + c][4][9]  (a,b,c,d rows of 9)
// ---------------------------------------------------------------------------
__global__ void spline_coeff_kernel(const float* __restrict__ ys,
                                    float* __restrict__ coef, int total)
{
    int t = blockIdx.x * blockDim.x + threadIdx.x;
    if (t >= total) return;                // total = B*E*3
    int c  = t % 3;
    int be = t / 3;
    int e  = be % 2;
    int b  = be / 2;

    float yt[10];
    #pragma unroll
    for (int j = 0; j < 10; ++j)
        yt[j] = ys[b * 60 + e * 30 + c * 10 + j] + (float)j * SPL_STEP;

    const float s6h2 = 6.0f * 81.0f;       // 6 / h^2, h = 1/9
    float rhs[8];
    #pragma unroll
    for (int i = 0; i < 8; ++i)
        rhs[i] = s6h2 * (yt[i] - 2.0f * yt[i + 1] + yt[i + 2]);

    float cp[8], dp[8];
    cp[0] = 0.25f; dp[0] = rhs[0] * 0.25f;
    #pragma unroll
    for (int i = 1; i < 8; ++i) {
        float m = 4.0f - cp[i - 1];
        cp[i] = 1.0f / m;
        dp[i] = (rhs[i] - dp[i - 1]) / m;
    }
    float M[10];
    M[0] = 0.0f; M[9] = 0.0f;
    M[8] = dp[7];
    #pragma unroll
    for (int i = 6; i >= 0; --i) M[i + 1] = dp[i] - cp[i] * M[i + 2];

    float* cq = coef + (size_t)t * 36;
    const float h = SPL_STEP;
    #pragma unroll
    for (int i = 0; i < 9; ++i) {
        cq[0 * 9 + i] = (M[i + 1] - M[i]) / (6.0f * h);
        cq[1 * 9 + i] = M[i] * 0.5f;
        cq[2 * 9 + i] = (yt[i + 1] - yt[i]) / h - (M[i + 1] + 2.0f * M[i]) * (h / 6.0f);
        cq[3 * 9 + i] = yt[i];
    }
}

// ---------------------------------------------------------------------------
// Streaming spline apply: read batch once (float4), write both E outputs.
// Coefficients for this (b,c) (both e) staged in LDS; prefetch next chunk.
// grid = B*3*chunks blocks; block = 256 threads x 4 pixels (chunk = 1024 px)
// ---------------------------------------------------------------------------
__global__ __launch_bounds__(256) void spline_apply_kernel(
    const float* __restrict__ batch, const float* __restrict__ coef,
    float* __restrict__ out, int Bn, int P, int chunks)
{
    __shared__ float lc[72];               // [e][q][i] = [2][4][9]
    int blk   = blockIdx.x;
    int chunk = blk % chunks;
    int bc    = blk / chunks;              // b*3 + c
    int b = bc / 3, c = bc - b * 3;

    if (threadIdx.x < 72) {
        int e = threadIdx.x / 36;
        int q = threadIdx.x - e * 36;
        lc[threadIdx.x] = coef[((size_t)(b * 2 + e) * 3 + c) * 36 + q];
    }
    __syncthreads();

    int p0 = chunk * 1024 + threadIdx.x * 4;
    if (p0 >= P) return;

    const float* src = batch + ((size_t)b * 3 + c) * P + p0;
    // speculative prefetch of the next chunk (global_prefetch_b8)
    __builtin_prefetch(src + 1024, 0, 1);

    const float4 xv4 = *(const float4*)src;
    float xs[4] = {xv4.x, xv4.y, xv4.z, xv4.w};
    float r0[4], r1[4];
    #pragma unroll
    for (int t = 0; t < 4; ++t) {
        float xv = xs[t];
        int xi = (int)floorf(xv * 9.0f);
        xi = xi < 0 ? 0 : (xi > 8 ? 8 : xi);
        float xf = xv - (float)xi * SPL_STEP;
        const float* c0 = lc;
        const float* c1 = lc + 36;
        r0[t] = ((c0[0*9+xi]*xf + c0[1*9+xi])*xf + c0[2*9+xi])*xf + c0[3*9+xi];
        r1[t] = ((c1[0*9+xi]*xf + c1[1*9+xi])*xf + c1[2*9+xi])*xf + c1[3*9+xi];
    }
    float4 v0 = {r0[0], r0[1], r0[2], r0[3]};
    float4 v1 = {r1[0], r1[1], r1[2], r1[3]};
    // out layout (E, B, 3, P)
    *(float4*)(out + (((size_t)b) * 3 + c) * P + p0)      = v0;
    *(float4*)(out + (((size_t)Bn + b) * 3 + c) * P + p0) = v1;
}

// ---------------------------------------------------------------------------
// 255-sample spline curves; output layout (E, B, 3, 255)
// ---------------------------------------------------------------------------
__global__ void spline_curve_kernel(const float* __restrict__ coef,
                                    float* __restrict__ out, int Bn)
{
    int t = blockIdx.x * blockDim.x + threadIdx.x;
    int total = Bn * 2 * 3 * 255;
    if (t >= total) return;
    int v  = t % 255;
    int r  = t / 255;
    int c  = r % 3;
    int be = r / 3;
    int e  = be % 2;
    int b  = be / 2;

    float val = (float)v / 255.0f;
    int vi = (int)floorf(val * 9.0f);
    vi = vi < 0 ? 0 : (vi > 8 ? 8 : vi);
    float vf = val - (float)vi * SPL_STEP;
    const float* cq = coef + ((size_t)(b * 2 + e) * 3 + c) * 36;
    float rv = ((cq[0*9+vi]*vf + cq[1*9+vi])*vf + cq[2*9+vi])*vf + cq[3*9+vi];
    out[(((size_t)e * Bn + b) * 3 + c) * 255 + v] = rv;
}

// ---------------------------------------------------------------------------
extern "C" void kernel_launch(void* const* d_in, const int* in_sizes, int n_in,
                              void* d_out, int out_size, void* d_ws, size_t ws_size,
                              hipStream_t stream)
{
    (void)in_sizes; (void)n_in; (void)out_size; (void)ws_size;

    const float* batch = (const float*)d_in[0];
    const float* w1 = (const float*)d_in[1];  const float* b1 = (const float*)d_in[2];
    const float* w2 = (const float*)d_in[3];  const float* b2 = (const float*)d_in[4];
    const float* w3 = (const float*)d_in[5];  const float* b3 = (const float*)d_in[6];
    const float* w4 = (const float*)d_in[7];  const float* b4 = (const float*)d_in[8];
    const float* w5 = (const float*)d_in[9];  const float* b5 = (const float*)d_in[10];
    const float* bn2[4] = {(const float*)d_in[11], (const float*)d_in[12],
                           (const float*)d_in[13], (const float*)d_in[14]};
    const float* bn3[4] = {(const float*)d_in[15], (const float*)d_in[16],
                           (const float*)d_in[17], (const float*)d_in[18]};
    const float* bn4[4] = {(const float*)d_in[19], (const float*)d_in[20],
                           (const float*)d_in[21], (const float*)d_in[22]};
    const float* bn5[4] = {(const float*)d_in[23], (const float*)d_in[24],
                           (const float*)d_in[25], (const float*)d_in[26]};
    const float* l1w = (const float*)d_in[27]; const float* l1b = (const float*)d_in[28];
    const float* l2w = (const float*)d_in[29]; const float* l2b = (const float*)d_in[30];

    float* ws = (float*)d_ws;
    size_t off = 0;
    float* y1   = ws + off; off += (size_t)BATCH * 8   * 127 * 127;
    float* y2   = ws + off; off += (size_t)BATCH * 16  * 63  * 63;
    float* y3   = ws + off; off += (size_t)BATCH * 32  * 31  * 31;
    float* y4   = ws + off; off += (size_t)BATCH * 64  * 15  * 15;
    float* y5   = ws + off; off += (size_t)BATCH * 128 * 7   * 7;
    float* feat = ws + off; off += (size_t)BATCH * 128;
    float* hh   = ws + off; off += (size_t)BATCH * 128;
    float* ysv  = ws + off; off += (size_t)BATCH * 60;
    float* coef = ws + off; off += (size_t)BATCH * 2 * 3 * 36;

    float* out_main  = (float*)d_out;
    float* out_curve = (float*)d_out + OUT_MAIN;

    auto conv = [&](const float* in, const float* w, const float* bias,
                    const float* const* bn, float* out,
                    int Cin, int Hin, int Cout, int Hout, int has_bn) {
        int S = Hout * Hout;
        int tiles = BATCH * ((S + 15) / 16) * ((Cout + 15) / 16);
        int blocks = (tiles + 7) / 8;          // 8 waves (256 threads) / block
        conv_wmma_kernel<<<blocks, 256, 0, stream>>>(
            in, w, bias,
            bn ? bn[0] : bias, bn ? bn[1] : bias,
            bn ? bn[2] : bias, bn ? bn[3] : bias,
            out, BATCH, Cin, Hin, Hin, Cout, Hout, Hout, has_bn);
    };

    conv(batch, w1, b1, nullptr, y1,  3, 256,   8, 127, 0);
    conv(y1,    w2, b2, bn2,     y2,  8, 127,  16,  63, 1);
    conv(y2,    w3, b3, bn3,     y3, 16,  63,  32,  31, 1);
    conv(y3,    w4, b4, bn4,     y4, 32,  31,  64,  15, 1);
    conv(y4,    w5, b5, bn5,     y5, 64,  15, 128,   7, 1);

    pool_kernel<<<(BATCH * 128 + 255) / 256, 256, 0, stream>>>(y5, feat, BATCH * 128, 49);

    // l1: 64x128 = relu(feat @ l1_w^T + b)   (tiles: 4x8 = 32 waves -> 4 blocks)
    fc_wmma_kernel<<<4, 256, 0, stream>>>(feat, l1w, l1b, hh, BATCH, 128, 128, 1);
    // l2: 64x60 = hh @ l2_w^T + b            (tiles: 4x4 = 16 waves -> 2 blocks)
    fc_wmma_kernel<<<2, 256, 0, stream>>>(hh, l2w, l2b, ysv, BATCH, 60, 128, 0);

    spline_coeff_kernel<<<(BATCH * 6 + 127) / 128, 128, 0, stream>>>(ysv, coef, BATCH * 6);

    const int chunks = PIX / 1024;             // 64
    spline_apply_kernel<<<BATCH * 3 * chunks, 256, 0, stream>>>(batch, coef, out_main,
                                                                BATCH, PIX, chunks);

    int curve_total = BATCH * 2 * 3 * 255;
    spline_curve_kernel<<<(curve_total + 255) / 256, 256, 0, stream>>>(coef, out_curve,
                                                                       BATCH);
}
